// loop_net_62474594288125
// MI455X (gfx1250) — compile-verified, matrix operand order checked by reading, not run on hardware
//
#include <hip/hip_runtime.h>
#include <hip/hip_bf16.h>

typedef __attribute__((ext_vector_type(16))) _Float16 v16h;
typedef __attribute__((ext_vector_type(8)))  _Float16 v8h;
typedef __attribute__((ext_vector_type(8)))  float    v8f;
typedef __attribute__((ext_vector_type(4)))  float    fvec4;   // plain vector, not HIP class

// explicit global-address-space pointers so B loads lower to global_load_b128
// (not flat_load_b128, which also ties up DScnt/LDS path)
typedef const __attribute__((address_space(1))) v8h*   gcv8h;
typedef const __attribute__((address_space(1))) fvec4* gcfvec4;

#define ITERS_  60
#define NTOT    61440      // B * ITERS = 1024 * 60
#define MTH_    4
#define DT_     0.01f

// ---------------------------------------------------------------------------
// helpers
// ---------------------------------------------------------------------------
__device__ __forceinline__ float leaky_act(float x) { return x >= 0.f ? x : 0.3f * x; }

__device__ __forceinline__ v8f wmma16(v16h a, v16h b, v8f c) {
    return __builtin_amdgcn_wmma_f32_16x16x32_f16(false, a, false, b, (short)0, c, false, false);
}

union frag_u { v16h v; v8h h[2]; };

// A fragment from LDS: 16 rows x 32 K, row-major f16, row stride strideH halves.
// lanes 0-15: row=lane, K 0..7 & 16..23 ; lanes 16-31: row=lane-16, K 8..15 & 24..31
__device__ __forceinline__ v16h load_fragA(const _Float16* base, int strideH, int kBase) {
    int lane = threadIdx.x & 31;
    const _Float16* p = base + (lane & 15) * strideH + kBase + ((lane & 16) ? 8 : 0);
    frag_u u;
    u.h[0] = *(const v8h*)p;
    u.h[1] = *(const v8h*)(p + 16);
    return u.v;
}

// B fragment from GLOBAL WT stored [dout][din] f16 (B^T): lane owns column
// n = nBase + (lane&15); same K split as A.
__device__ __forceinline__ v16h load_fragB(const _Float16* WT, int strideH, int nBase, int kBase) {
    int lane = threadIdx.x & 31;
    const _Float16* p = WT + (size_t)(nBase + (lane & 15)) * strideH + kBase + ((lane & 16) ? 8 : 0);
    gcv8h g = (gcv8h)p;
    frag_u u;
    u.h[0] = g[0];
    u.h[1] = g[2];   // +16 halves
    return u.v;
}

// C layout: vgpr r, lanes 0-15 -> (M=r, N=lane) ; lanes 16-31 -> (M=r+8, N=lane-16)
__device__ __forceinline__ void storeC_f16_leaky(v8f c, _Float16* dst, int strideH, int nBase,
                                                 const float* bvec) {
    int lane = threadIdx.x & 31;
    int col = nBase + (lane & 15);
    int rowOff = (lane & 16) ? 8 : 0;
    float bias = bvec[col];
#pragma unroll
    for (int r = 0; r < 8; ++r)
        dst[(rowOff + r) * strideH + col] = (_Float16)leaky_act(c[r] + bias);
}

__device__ __forceinline__ void storeC_f16_tanh(v8f c, _Float16* dst, int strideH, int nBase,
                                                float biasLane) {
    int lane = threadIdx.x & 31;
    int col = nBase + (lane & 15);
    int rowOff = (lane & 16) ? 8 : 0;
#pragma unroll
    for (int r = 0; r < 8; ++r)
        dst[(rowOff + r) * strideH + col] = (_Float16)tanhf(c[r] + biasLane);
}

__device__ __forceinline__ void storeC_f32_leaky(v8f c, float* dst, int strideF, int rowBase,
                                                 int nBase, const float* bvec) {
    int lane = threadIdx.x & 31;
    int col = nBase + (lane & 15);
    int rowOff = (lane & 16) ? 8 : 0;
    float bias = bvec[col];
#pragma unroll
    for (int r = 0; r < 8; ++r)
        dst[(size_t)(rowBase + rowOff + r) * strideF + col] = leaky_act(c[r] + bias);
}

// ---------------------------------------------------------------------------
// weight conversion: dst[n*dinPad + k] = (k < din) ? src[k*dout + n] : 0   (f32 -> f16)
// ---------------------------------------------------------------------------
__global__ void convT_kernel(const float* __restrict__ src, _Float16* __restrict__ dst,
                             int din, int dout, int dinPad) {
    int idx = blockIdx.x * 256 + threadIdx.x;
    if (idx >= dout * dinPad) return;
    int n = idx / dinPad, k = idx % dinPad;
    dst[idx] = (k < din) ? (_Float16)src[k * dout + n] : (_Float16)0.f;
}

// aux weights: src [nNets][din][32] -> dst[net*netStride + n*32 + k] (K padded to 32)
__global__ void convTaux_kernel(const float* __restrict__ src, _Float16* __restrict__ dst,
                                int din, int nNets, int netStride) {
    int idx = blockIdx.x * 256 + threadIdx.x;
    if (idx >= nNets * 1024) return;
    int net = idx >> 10, rem = idx & 1023, n = rem >> 5, k = rem & 31;
    dst[net * netStride + n * 32 + k] =
        (k < din) ? (_Float16)src[net * din * 32 + k * 32 + n] : (_Float16)0.f;
}

// ---------------------------------------------------------------------------
// shared 256->256 layer: 64-row tile, wave owns cols [32w,32w+32), 4 row sub-tiles.
// Per k-step: 2 global B loads + 4 LDS A loads (all preloaded) feed 8 WMMAs.
// ---------------------------------------------------------------------------
__device__ __forceinline__ void mlp_layer256(_Float16 (&Abuf)[64][256],
                                             const _Float16* __restrict__ WT,
                                             const float* __restrict__ bvec, int wave) {
    int nBase = wave * 32;
    v8f c[4][2] = {};
    for (int k = 0; k < 256; k += 32) {
        v16h B0 = load_fragB(WT, 256, nBase, k);
        v16h B1 = load_fragB(WT, 256, nBase + 16, k);
        v16h a[4];
#pragma unroll
        for (int rt = 0; rt < 4; ++rt) a[rt] = load_fragA(&Abuf[rt * 16][0], 256, k);
#pragma unroll
        for (int rt = 0; rt < 4; ++rt) {
            c[rt][0] = wmma16(a[rt], B0, c[rt][0]);
            c[rt][1] = wmma16(a[rt], B1, c[rt][1]);
        }
    }
    __syncthreads();
#pragma unroll
    for (int rt = 0; rt < 4; ++rt) {
        storeC_f16_leaky(c[rt][0], &Abuf[rt * 16][0], 256, nBase, bvec);
        storeC_f16_leaky(c[rt][1], &Abuf[rt * 16][0], 256, nBase + 16, bvec);
    }
    __syncthreads();
}

// ---------------------------------------------------------------------------
// encoder: 256 -> 256 -> 256 -> 16, leaky each layer.  block = 256 thr (8 waves),
// 64-row tile per block.
// ---------------------------------------------------------------------------
__global__ void enc_kernel(const float* __restrict__ inp,
                           const _Float16* __restrict__ W1T, const float* __restrict__ b1,
                           const _Float16* __restrict__ W2T, const float* __restrict__ b2,
                           const _Float16* __restrict__ W3T, const float* __restrict__ b3,
                           float* __restrict__ emb) {
    __shared__ __align__(16) _Float16 Abuf[64][256];
    int rowBase = blockIdx.x * 64;
    int tid = threadIdx.x, wave = tid >> 5;

    { // load 64x256 fp32 tile (slice [:, :60, :] of (B,64,256)) -> f16 LDS
        int r = tid >> 2;              // 0..63
        int c0 = (tid & 3) * 64;       // 0,64,128,192
        int sample = rowBase + r;
        int bb = sample / ITERS_, tq = sample % ITERS_;
        const float* src = inp + ((size_t)(bb * 64 + tq)) * 256 + c0;
#pragma unroll
        for (int i = 0; i < 64; i += 4) {
            fvec4 v = *(gcfvec4)(src + i);
            Abuf[r][c0 + i + 0] = (_Float16)v[0];
            Abuf[r][c0 + i + 1] = (_Float16)v[1];
            Abuf[r][c0 + i + 2] = (_Float16)v[2];
            Abuf[r][c0 + i + 3] = (_Float16)v[3];
        }
    }
    __syncthreads();

    mlp_layer256(Abuf, W1T, b1, wave);
    mlp_layer256(Abuf, W2T, b2, wave);

    if (wave < 4) { // 256 -> 16 : wave w handles row sub-tile w
        v8f c = {};
        for (int k = 0; k < 256; k += 32) {
            v16h a = load_fragA(&Abuf[wave * 16][0], 256, k);
            c = wmma16(a, load_fragB(W3T, 256, 0, k), c);
        }
        storeC_f32_leaky(c, emb, 16, rowBase + wave * 16, 0, b3);
    }
}

// ---------------------------------------------------------------------------
// Koopman scan: block = 320 thr (10 waves) = one wave per eigen-net,
// one 16-row tile; MTH_ iterations fully fused in LDS.
// ---------------------------------------------------------------------------
__global__ void koop_kernel(const float* __restrict__ emb, float* __restrict__ yfin,
                            const _Float16* __restrict__ WxTr, const _Float16* __restrict__ WhTr,
                            const _Float16* __restrict__ WxTc, const _Float16* __restrict__ WhTc,
                            const float* __restrict__ br0, const float* __restrict__ br1,
                            const float* __restrict__ br2,
                            const float* __restrict__ bc0, const float* __restrict__ bc1,
                            const float* __restrict__ bc2,
                            const float* __restrict__ WoR, const float* __restrict__ boR,
                            const float* __restrict__ WoC, const float* __restrict__ boC) {
    __shared__ __align__(16) float    yf[16][16];
    __shared__ __align__(16) _Float16 yA[16][32];
    __shared__ __align__(16) _Float16 st[10][3][16][32];
    __shared__ __align__(16) float    om[16][16];
    int tid = threadIdx.x, wave = tid >> 5, lane = tid & 31;
    int rowBase = blockIdx.x * 16;

    for (int i = tid; i < 256; i += 320) {
        int r = i >> 4, c = i & 15;
        float v = emb[(size_t)(rowBase + r) * 16 + c];
        yf[r][c] = v; yA[r][c] = (_Float16)v; yA[r][c + 16] = (_Float16)0.f;
    }
    for (int i = tid; i < 10 * 3 * 16 * 32; i += 320) (&st[0][0][0][0])[i] = (_Float16)0.f;
    __syncthreads();

    int net = wave;                       // 0..3 real, 4..9 complex
    bool isReal = net < 4;
    const _Float16* Wx = isReal ? WxTr + net * 3072 : WxTc + (net - 4) * 3072;
    const _Float16* Wh = isReal ? WhTr + net * 3072 : WhTc + (net - 4) * 3072;
    const float* brs[3] = {br0, br1, br2};
    const float* bcs[3] = {bc0, bc1, bc2};

    v16h Bx[3][2], Bh[3][2];
    float biasL[3][2];
#pragma unroll
    for (int l = 0; l < 3; ++l) {
        const float* bl = isReal ? brs[l] + net * 32 : bcs[l] + (net - 4) * 32;
#pragma unroll
        for (int t = 0; t < 2; ++t) {
            Bx[l][t] = load_fragB(Wx + l * 1024, 32, t * 16, 0);
            Bh[l][t] = load_fragB(Wh + l * 1024, 32, t * 16, 0);
            biasL[l][t] = bl[t * 16 + (lane & 15)];
        }
    }

    for (int it = 0; it < MTH_; ++it) {
        const _Float16* inpPtr = &yA[0][0];
#pragma unroll
        for (int l = 0; l < 3; ++l) {
            v16h aIn = load_fragA(inpPtr, 32, 0);
            v16h aSt = load_fragA(&st[net][l][0][0], 32, 0);
            v8f c0 = {}, c1 = {};
            c0 = wmma16(aIn, Bx[l][0], c0); c0 = wmma16(aSt, Bh[l][0], c0);
            c1 = wmma16(aIn, Bx[l][1], c1); c1 = wmma16(aSt, Bh[l][1], c1);
            storeC_f16_tanh(c0, &st[net][l][0][0], 32, 0,  biasL[l][0]);
            storeC_f16_tanh(c1, &st[net][l][0][0], 32, 16, biasL[l][1]);
            inpPtr = &st[net][l][0][0];
        }
        // omega heads: per-lane 32-MAC dot on final hidden state
        if (isReal) {
            if (lane < 16) {
                int s = lane;
                const float* wo = WoR + net * 32;
                float acc = boR[net];
#pragma unroll
                for (int k = 0; k < 32; ++k) acc += (float)st[net][2][s][k] * wo[k];
                om[s][net] = acc;
            }
        } else {
            int s = lane & 15, o = lane >> 4;
            const float* wo = WoC + (net - 4) * 64;
            float acc = boC[(net - 4) * 2 + o];
#pragma unroll
            for (int k = 0; k < 32; ++k) acc += (float)st[net][2][s][k] * wo[k * 2 + o];
            om[s][4 + (net - 4) * 2 + o] = acc;
        }
        __syncthreads();
        if (tid < 160) { // eigen update: 16 samples x 10 modes
            int s = tid & 15, mode = tid >> 4;
            if (mode < 4) {
                float v = yf[s][mode] * __expf(om[s][mode] * DT_);
                yf[s][mode] = v; yA[s][mode] = (_Float16)v;
            } else {
                int m = mode - 4;
                float mu = om[s][4 + 2 * m], w = om[s][5 + 2 * m];
                float y1 = yf[s][4 + 2 * m], y2 = yf[s][5 + 2 * m];
                float sc = __expf(mu * DT_);
                float cc = __cosf(w * DT_), sn = __sinf(w * DT_);
                float n1 = sc * (cc * y1 - sn * y2);
                float n2 = sc * (sn * y1 + cc * y2);
                yf[s][4 + 2 * m] = n1; yf[s][5 + 2 * m] = n2;
                yA[s][4 + 2 * m] = (_Float16)n1; yA[s][5 + 2 * m] = (_Float16)n2;
            }
        }
        __syncthreads();
    }

    for (int i = tid; i < 256; i += 320) {
        int r = i >> 4, c = i & 15;
        yfin[(size_t)(rowBase + r) * 16 + c] = yf[r][c];
    }
}

// ---------------------------------------------------------------------------
// decoder: 16 -> 256 -> 256 -> 256, leaky each layer, fp32 out. 64-row tile.
// ---------------------------------------------------------------------------
__global__ void dec_kernel(const float* __restrict__ yfin,
                           const _Float16* __restrict__ W1T, const float* __restrict__ b1,
                           const _Float16* __restrict__ W2T, const float* __restrict__ b2,
                           const _Float16* __restrict__ W3T, const float* __restrict__ b3,
                           float* __restrict__ out) {
    __shared__ __align__(16) _Float16 A0[64][32];
    __shared__ __align__(16) _Float16 Abuf[64][256];
    int rowBase = blockIdx.x * 64;
    int tid = threadIdx.x, wave = tid >> 5;

    for (int i = tid; i < 1024; i += 256) { // latent tile, K padded 16 -> 32
        int r = i >> 4, c = i & 15;
        A0[r][c] = (_Float16)yfin[(size_t)(rowBase + r) * 16 + c];
        A0[r][c + 16] = (_Float16)0.f;
    }
    __syncthreads();

    { // layer 1: K=32 (padded), 256 cols
        int nBase = wave * 32;
        v16h B0 = load_fragB(W1T, 32, nBase, 0);
        v16h B1 = load_fragB(W1T, 32, nBase + 16, 0);
#pragma unroll
        for (int rt = 0; rt < 4; ++rt) {
            v16h a = load_fragA(&A0[rt * 16][0], 32, 0);
            v8f c0 = {}, c1 = {};
            c0 = wmma16(a, B0, c0);
            c1 = wmma16(a, B1, c1);
            storeC_f16_leaky(c0, &Abuf[rt * 16][0], 256, nBase, b1);
            storeC_f16_leaky(c1, &Abuf[rt * 16][0], 256, nBase + 16, b1);
        }
    }
    __syncthreads();

    mlp_layer256(Abuf, W2T, b2, wave);

    { // layer 3: 256 -> 256, fp32 to global
        int nBase = wave * 32;
        v8f c[4][2] = {};
        for (int k = 0; k < 256; k += 32) {
            v16h B0 = load_fragB(W3T, 256, nBase, k);
            v16h B1 = load_fragB(W3T, 256, nBase + 16, k);
            v16h a[4];
#pragma unroll
            for (int rt = 0; rt < 4; ++rt) a[rt] = load_fragA(&Abuf[rt * 16][0], 256, k);
#pragma unroll
            for (int rt = 0; rt < 4; ++rt) {
                c[rt][0] = wmma16(a[rt], B0, c[rt][0]);
                c[rt][1] = wmma16(a[rt], B1, c[rt][1]);
            }
        }
#pragma unroll
        for (int rt = 0; rt < 4; ++rt) {
            storeC_f32_leaky(c[rt][0], out, 256, rowBase + rt * 16, nBase, b3);
            storeC_f32_leaky(c[rt][1], out, 256, rowBase + rt * 16, nBase + 16, b3);
        }
    }
}

// ---------------------------------------------------------------------------
// launch
// ---------------------------------------------------------------------------
extern "C" void kernel_launch(void* const* d_in, const int* in_sizes, int n_in,
                              void* d_out, int out_size, void* d_ws, size_t ws_size,
                              hipStream_t stream) {
    (void)in_sizes; (void)n_in; (void)out_size; (void)ws_size;
    // d_in order (setup_inputs dict order, pytree leaves):
    // 0 inputs | 1-6 enc (W,b)x3 | 7-12 dec (W,b)x3
    // 13-23 aux_r: Wx0,Wx1,Wx2,Wh0,Wh1,Wh2,b0,b1,b2,Wo,bo
    // 24-34 aux_c: same
    const float* inp = (const float*)d_in[0];
    const float* encW[3] = {(const float*)d_in[1], (const float*)d_in[3], (const float*)d_in[5]};
    const float* encB[3] = {(const float*)d_in[2], (const float*)d_in[4], (const float*)d_in[6]};
    const float* decW[3] = {(const float*)d_in[7], (const float*)d_in[9], (const float*)d_in[11]};
    const float* decB[3] = {(const float*)d_in[8], (const float*)d_in[10], (const float*)d_in[12]};

    char* w = (char*)d_ws;
    _Float16* encW1T = (_Float16*)w; w += 65536 * 2;
    _Float16* encW2T = (_Float16*)w; w += 65536 * 2;
    _Float16* encW3T = (_Float16*)w; w += 4096 * 2;    // [16][256]
    _Float16* decW1T = (_Float16*)w; w += 8192 * 2;    // [256][32] K-padded
    _Float16* decW2T = (_Float16*)w; w += 65536 * 2;
    _Float16* decW3T = (_Float16*)w; w += 65536 * 2;
    _Float16* WxTr   = (_Float16*)w; w += 4 * 3 * 1024 * 2;
    _Float16* WhTr   = (_Float16*)w; w += 4 * 3 * 1024 * 2;
    _Float16* WxTc   = (_Float16*)w; w += 6 * 3 * 1024 * 2;
    _Float16* WhTc   = (_Float16*)w; w += 6 * 3 * 1024 * 2;
    float* emb  = (float*)w; w += (size_t)NTOT * 16 * 4;
    float* yfin = (float*)w; w += (size_t)NTOT * 16 * 4;

    convT_kernel<<<256, 256, 0, stream>>>(encW[0], encW1T, 256, 256, 256);
    convT_kernel<<<256, 256, 0, stream>>>(encW[1], encW2T, 256, 256, 256);
    convT_kernel<<<16, 256, 0, stream>>>(encW[2], encW3T, 256, 16, 256);
    convT_kernel<<<32, 256, 0, stream>>>(decW[0], decW1T, 16, 256, 32);
    convT_kernel<<<256, 256, 0, stream>>>(decW[1], decW2T, 256, 256, 256);
    convT_kernel<<<256, 256, 0, stream>>>(decW[2], decW3T, 256, 256, 256);
    for (int l = 0; l < 3; ++l) {
        int dinX = (l == 0) ? 16 : 32;
        convTaux_kernel<<<16, 256, 0, stream>>>((const float*)d_in[13 + l], WxTr + l * 1024,
                                                dinX, 4, 3072);
        convTaux_kernel<<<16, 256, 0, stream>>>((const float*)d_in[16 + l], WhTr + l * 1024,
                                                32, 4, 3072);
        convTaux_kernel<<<24, 256, 0, stream>>>((const float*)d_in[24 + l], WxTc + l * 1024,
                                                dinX, 6, 3072);
        convTaux_kernel<<<24, 256, 0, stream>>>((const float*)d_in[27 + l], WhTc + l * 1024,
                                                32, 6, 3072);
    }

    enc_kernel<<<NTOT / 64, 256, 0, stream>>>(inp, encW1T, encB[0], encW2T, encB[1],
                                              encW3T, encB[2], emb);
    koop_kernel<<<NTOT / 16, 320, 0, stream>>>(emb, yfin, WxTr, WhTr, WxTc, WhTc,
                                               (const float*)d_in[19], (const float*)d_in[20],
                                               (const float*)d_in[21],
                                               (const float*)d_in[30], (const float*)d_in[31],
                                               (const float*)d_in[32],
                                               (const float*)d_in[22], (const float*)d_in[23],
                                               (const float*)d_in[33], (const float*)d_in[34]);
    dec_kernel<<<NTOT / 64, 256, 0, stream>>>(yfin, decW1T, decB[0], decW2T, decB[1],
                                              decW3T, decB[2], (float*)d_out);
}
